// MultiScaleRetention_41291815584172
// MI455X (gfx1250) — matrix operand order repaired and checked by reading, not compiled, not run on hardware
//
#include <hip/hip_runtime.h>
#include <hip/hip_fp16.h>

// MultiScaleRetention on gfx1250 (MI455X), wave32 + v_wmma_f32_16x16x32_f16.
// B=2, T=2048, C=1024, H=16, DH=64.
// All WMMA inputs are O(1) -> fp16 (e5m10) beats bf16 precision at identical
// 16x16x32 matrix-pipe throughput; accumulation stays f32.

#define BB 2
#define TT 2048
#define CC 1024
#define HH 16
#define DH 64

typedef __attribute__((ext_vector_type(16))) _Float16 v16h;
typedef __attribute__((ext_vector_type(8)))  float    v8f;

__device__ __forceinline__ v8f wmma_f16(v16h a, v16h b, v8f c) {
  // 8 args: (neg_a, A, neg_b, B, c_mod, C, reuse_a, reuse_b)
  return __builtin_amdgcn_wmma_f32_16x16x32_f16(false, a, false, b, (short)0, c,
                                                false, false);
}

// ---- A fragment (16x32, MxK), row-major source. Per CDNA5 ISA:
// lane m = lane&15; VGPR v: K = 16*(v/4) + 8*(lane/16) + 2*(v%4), pair {K,K+1}.
__device__ __forceinline__ v16h load_a_f32(const float* __restrict__ A, int lda,
                                           int row0, int k0, int lane) {
  const int m = lane & 15, g = lane >> 4;
  const float* Ar = A + (size_t)(row0 + m) * lda + k0;
  v16h a;
#pragma unroll
  for (int v = 0; v < 8; ++v) {
    const int kk = ((v >> 2) << 4) + (g << 3) + ((v & 3) << 1);
    a[2 * v]     = (_Float16)Ar[kk];
    a[2 * v + 1] = (_Float16)Ar[kk + 1];
  }
  return a;
}

__device__ __forceinline__ v16h load_a_f16(const _Float16* __restrict__ A, int lda,
                                           int row0, int k0, int lane) {
  const int m = lane & 15, g = lane >> 4;
  const _Float16* Ar = A + (size_t)(row0 + m) * lda + k0;
  v16h a;
#pragma unroll
  for (int v = 0; v < 8; ++v) {
    const int kk = ((v >> 2) << 4) + (g << 3) + ((v & 3) << 1);
    a[2 * v]     = Ar[kk];
    a[2 * v + 1] = Ar[kk + 1];
  }
  return a;
}

// ---- B fragment (32x16, KxN). lane n = lane&15; element e: K = 8*(lane/16)+e+(e&8).
// Source indexed as Bsrc[(col0+n)*ldb + k0 + K]  (i.e. B[k][n] = Bsrc[n][k]).
__device__ __forceinline__ v16h load_b_f32(const float* __restrict__ Bsrc, int ldb,
                                           int col0, int k0, int lane) {
  const int n = lane & 15, g = lane >> 4;
  const float* Br = Bsrc + (size_t)(col0 + n) * ldb + k0 + (g << 3);
  v16h b;
#pragma unroll
  for (int e = 0; e < 16; ++e) b[e] = (_Float16)Br[e + (e & 8)];
  return b;
}

__device__ __forceinline__ v16h load_b_f16(const _Float16* __restrict__ Bsrc, int ldb,
                                           int col0, int k0, int lane) {
  const int n = lane & 15, g = lane >> 4;
  const _Float16* Br = Bsrc + (size_t)(col0 + n) * ldb + k0 + (g << 3);
  v16h b;
#pragma unroll
  for (int e = 0; e < 16; ++e) b[e] = Br[e + (e & 8)];
  return b;
}

// ---- Kernel 1: fused Q/K/V projection (+RoPE for Q,K; transpose-store for V).
// Each wave owns the column pair (d, d+32) of one head so RoPE pairs stay in-lane.
__global__ void __launch_bounds__(256) msr_proj_rope(
    const float* __restrict__ x, const float* __restrict__ Wq,
    const float* __restrict__ Wk, const float* __restrict__ Wv,
    _Float16* __restrict__ qws, _Float16* __restrict__ kws,
    _Float16* __restrict__ vtws) {
  const int lane = threadIdx.x & 31;
  const int wid  = blockIdx.x * (blockDim.x >> 5) + (threadIdx.x >> 5);
  const int dt  = wid & 1;          // which 16-wide slice of d in [0,32)
  const int h   = (wid >> 1) & 15;
  const int mt  = (wid >> 5) & 255; // row tile over B*T = 4096
  const int sel = wid >> 13;        // 0=q 1=k 2=v
  if (sel > 2) return;
  const float* W = (sel == 0) ? Wq : ((sel == 1) ? Wk : Wv);
  const int row0 = mt * 16;
  const int col0 = h * DH + dt * 16;

  v8f acc0 = {}, acc1 = {};
  for (int k0 = 0; k0 < CC; k0 += 32) {
    v16h a  = load_a_f32(x, CC, row0, k0, lane);
    v16h b0 = load_b_f32(W, CC, col0, k0, lane);        // cols d in [dt*16, dt*16+16)
    v16h b1 = load_b_f32(W, CC, col0 + 32, k0, lane);   // cols d+32
    acc0 = wmma_f16(a, b0, acc0);
    acc1 = wmma_f16(a, b1, acc1);
  }

  const int g = lane >> 4, n = lane & 15;
  const int d0 = dt * 16 + n;                            // in [0,32)
  // inv_freq = 10000^(-d0/32) = exp2(-d0 * log2(10000)/32)
  const float inv = exp2f(-(float)d0 * (13.287712379549449f / 32.0f));
#pragma unroll
  for (int r = 0; r < 8; ++r) {
    const int row = row0 + r + 8 * g;  // b*T + t
    const int b   = row >> 11;
    const int t   = row & (TT - 1);
    const float v0 = acc0[r], v1 = acc1[r];
    float o0, o1;
    if (sel < 2) {
      float sn, cs;
      __sincosf((float)t * inv, &sn, &cs);  // hw v_sin/v_cos path; bf16-level accuracy
      o0 = v0 * cs - v1 * sn;  // d < 32: x1*c - x2*s
      o1 = v1 * cs + v0 * sn;  // d >= 32: x2*c + x1*s
    } else {
      o0 = v0; o1 = v1;
    }
    if (sel == 2) {
      // v transposed: vT[b][h][d][t]
      const size_t base = (((size_t)b * HH + h) * DH) * TT;
      vtws[base + (size_t)d0 * TT + t]        = (_Float16)o0;
      vtws[base + (size_t)(d0 + 32) * TT + t] = (_Float16)o1;
    } else {
      _Float16* dst = (sel == 0) ? qws : kws;
      const size_t base = (((size_t)b * HH + h) * TT + t) * DH;
      dst[base + d0]      = (_Float16)o0;
      dst[base + d0 + 32] = (_Float16)o1;
    }
  }
}

// ---- Kernel 2: retention attention + GroupNorm.
// One wave per (b, h, 16-row t-tile). s processed in 32-wide chunks:
// S = (Q Kt)*scale*decay  ->  f16 A-frag via 1KB LDS relayout  ->  O += S*V.
__global__ void __launch_bounds__(256) msr_retention(
    const _Float16* __restrict__ qws, const _Float16* __restrict__ kws,
    const _Float16* __restrict__ vtws, const float* __restrict__ gnw,
    const float* __restrict__ gnb, _Float16* __restrict__ yws) {
  __shared__ _Float16 Pbuf[8][16 * 32];
  const int lane = threadIdx.x & 31;
  const int widx = threadIdx.x >> 5;
  const int wid  = blockIdx.x * 8 + widx;
  const int tt = wid & 127;         // t tile (T/16 = 128)
  const int h  = (wid >> 7) & 15;
  const int b  = wid >> 11;
  const int t0 = tt * 16;
  const _Float16* qb = qws  + (((size_t)b * HH + h) * TT) * DH;
  const _Float16* kb = kws  + (((size_t)b * HH + h) * TT) * DH;
  const _Float16* vt = vtws + (((size_t)b * HH + h) * DH) * TT;
  const int g = lane >> 4, n = lane & 15;

  const float gamma = 1.0f - exp2f(-5.0f - (float)h);
  const float lg    = log2f(gamma);
  const float scale = 0.125f;  // 1/sqrt(64)

  const v16h aq0 = load_a_f16(qb, DH, t0, 0, lane);
  const v16h aq1 = load_a_f16(qb, DH, t0, 32, lane);
  v8f o0 = {}, o1 = {}, o2 = {}, o3 = {};
  _Float16* P = &Pbuf[widx][0];

  for (int s0 = 0; s0 < t0 + 16; s0 += 32) {
    // QK^T for two 16-wide s tiles (B-frag: lanes = s, K = d contiguous in q/k layout)
    v8f S0 = {}, S1 = {};
    S0 = wmma_f16(aq0, load_b_f16(kb, DH, s0, 0, lane), S0);
    S0 = wmma_f16(aq1, load_b_f16(kb, DH, s0, 32, lane), S0);
    S1 = wmma_f16(aq0, load_b_f16(kb, DH, s0 + 16, 0, lane), S1);
    S1 = wmma_f16(aq1, load_b_f16(kb, DH, s0 + 16, 32, lane), S1);
    // decay * scale in accumulator layout, pack to LDS as A-fragment source
#pragma unroll
    for (int r = 0; r < 8; ++r) {
      const int m    = r + 8 * g;
      const int trow = t0 + m;
      const int dA = trow - (s0 + n);
      const int dBv = dA - 16;
      const float p0 = (dA  >= 0) ? S0[r] * scale * exp2f(lg * (float)dA)  : 0.0f;
      const float p1 = (dBv >= 0) ? S1[r] * scale * exp2f(lg * (float)dBv) : 0.0f;
      P[m * 32 + n]      = (_Float16)p0;
      P[m * 32 + 16 + n] = (_Float16)p1;
    }
    __builtin_amdgcn_wave_barrier();  // same-wave DS ops are in-order; pin scheduling
    // reload in A-fragment pattern (16x32, K = s offset)
    v16h aP;
    {
      const int m = lane & 15;
#pragma unroll
      for (int v = 0; v < 8; ++v) {
        const int kk = ((v >> 2) << 4) + (g << 3) + ((v & 3) << 1);
        aP[2 * v]     = P[m * 32 + kk];
        aP[2 * v + 1] = P[m * 32 + kk + 1];
      }
    }
    // O += P * V  (B-frag from vT: lanes = d, K = s contiguous)
    o0 = wmma_f16(aP, load_b_f16(vt, TT, 0,  s0, lane), o0);
    o1 = wmma_f16(aP, load_b_f16(vt, TT, 16, s0, lane), o1);
    o2 = wmma_f16(aP, load_b_f16(vt, TT, 32, s0, lane), o2);
    o3 = wmma_f16(aP, load_b_f16(vt, TT, 48, s0, lane), o3);
    __builtin_amdgcn_wave_barrier();
  }

  // GroupNorm over DH=64: each row lives in one 16-lane half-group x 4 tiles.
  const float w0 = gnw[h * DH + n],      w1 = gnw[h * DH + 16 + n];
  const float w2 = gnw[h * DH + 32 + n], w3 = gnw[h * DH + 48 + n];
  const float c0 = gnb[h * DH + n],      c1 = gnb[h * DH + 16 + n];
  const float c2 = gnb[h * DH + 32 + n], c3 = gnb[h * DH + 48 + n];
#pragma unroll
  for (int r = 0; r < 8; ++r) {
    const float x0 = o0[r], x1 = o1[r], x2 = o2[r], x3 = o3[r];
    float sum = x0 + x1 + x2 + x3;
    float sq  = x0 * x0 + x1 * x1 + x2 * x2 + x3 * x3;
#pragma unroll
    for (int mask = 1; mask <= 8; mask <<= 1) {  // stays inside 16-lane group
      sum += __shfl_xor(sum, mask, 32);
      sq  += __shfl_xor(sq, mask, 32);
    }
    const float mean = sum * (1.0f / 64.0f);
    const float var  = sq * (1.0f / 64.0f) - mean * mean;
    const float rstd = rsqrtf(var + 1e-5f);
    const int trow = t0 + r + 8 * g;
    const size_t base = ((size_t)b * TT + trow) * CC + h * DH;
    yws[base + n]      = (_Float16)((x0 - mean) * rstd * w0 + c0);
    yws[base + 16 + n] = (_Float16)((x1 - mean) * rstd * w1 + c1);
    yws[base + 32 + n] = (_Float16)((x2 - mean) * rstd * w2 + c2);
    yws[base + 48 + n] = (_Float16)((x3 - mean) * rstd * w3 + c3);
  }
}

// ---- Kernel 3: final projection y @ Wo^T -> f32 out.
__global__ void __launch_bounds__(256) msr_outproj(
    const _Float16* __restrict__ y, const float* __restrict__ Wo,
    float* __restrict__ out) {
  const int lane = threadIdx.x & 31;
  const int wid  = blockIdx.x * 8 + (threadIdx.x >> 5);
  const int nt = wid & 63;
  const int mt = wid >> 6;
  const int row0 = mt * 16, col0 = nt * 16;
  v8f acc = {};
  for (int k0 = 0; k0 < CC; k0 += 32) {
    v16h a  = load_a_f16(y, CC, row0, k0, lane);
    v16h bw = load_b_f32(Wo, CC, col0, k0, lane);
    acc = wmma_f16(a, bw, acc);
  }
  const int g = lane >> 4, n = lane & 15;
#pragma unroll
  for (int r = 0; r < 8; ++r) {
    const int row = row0 + r + 8 * g;
    out[(size_t)row * CC + col0 + n] = acc[r];
  }
}

extern "C" void kernel_launch(void* const* d_in, const int* in_sizes, int n_in,
                              void* d_out, int out_size, void* d_ws, size_t ws_size,
                              hipStream_t stream) {
  const float* x   = (const float*)d_in[0];
  const float* Wq  = (const float*)d_in[1];
  const float* Wk  = (const float*)d_in[2];
  const float* Wv  = (const float*)d_in[3];
  const float* Wo  = (const float*)d_in[4];
  const float* gnw = (const float*)d_in[5];
  const float* gnb = (const float*)d_in[6];
  float* out = (float*)d_out;

  const size_t qkv_bytes = (size_t)BB * HH * TT * DH * sizeof(_Float16);  // 8 MB each
  char* ws = (char*)d_ws;
  _Float16* qws  = (_Float16*)(ws);
  _Float16* kws  = (_Float16*)(ws + qkv_bytes);
  _Float16* vtws = (_Float16*)(ws + 2 * qkv_bytes);
  _Float16* yws  = (_Float16*)(ws + 3 * qkv_bytes);

  // 3 GEMMs x (256 row tiles * 16 heads * 2 d-slices) waves, 8 waves/block
  msr_proj_rope<<<3072, 256, 0, stream>>>(x, Wq, Wk, Wv, qws, kws, vtws);
  // B*H*(T/16) = 4096 wave tiles
  msr_retention<<<512, 256, 0, stream>>>(qws, kws, vtws, gnw, gnb, yws);
  // 256 * 64 = 16384 wave tiles
  msr_outproj<<<2048, 256, 0, stream>>>(yws, Wo, out);
}
